// GNN_QFM_79740362818136
// MI455X (gfx1250) — matrix operand order repaired and checked by reading, not compiled, not run on hardware
//
#include <hip/hip_runtime.h>
#include <hip/hip_bf16.h>

// ---------------------------------------------------------------------------
// Problem constants (from reference)
// ---------------------------------------------------------------------------
#define NN    16384
#define EE    262144
#define FF    16
#define HH    128
#define HEADS 4
#define LL    5
#define GG    128
#define DFF   512
#define EPSB  1e-5f

typedef __attribute__((ext_vector_type(16))) _Float16 v16h;
typedef __attribute__((ext_vector_type(8)))  _Float16 v8h;
typedef __attribute__((ext_vector_type(8)))  float    v8f;

// ---------------------------------------------------------------------------
// float <-> ordered uint key (monotonic); max is order-invariant -> atomicMax ok
// ---------------------------------------------------------------------------
__device__ __forceinline__ unsigned f2key(float f) {
    int i = __float_as_int(f);
    return (i >= 0) ? ((unsigned)i | 0x80000000u) : ~(unsigned)i;
}
__device__ __forceinline__ float key2f(unsigned k) {
    int i = (k & 0x80000000u) ? (int)(k & 0x7fffffffu) : (int)~k;
    return __int_as_float(i);
}

// ---------------------------------------------------------------------------
// Fill kernels
// ---------------------------------------------------------------------------
__global__ void fill_u32_kernel(unsigned* __restrict__ p, unsigned v, int n) {
    int i = blockIdx.x * blockDim.x + threadIdx.x;
    if (i < n) p[i] = v;
}

// ---------------------------------------------------------------------------
// Embedding GEMM: t[N,128] = x[N,16] @ embW[16,128] + embb (K=16 -> scalar VALU)
// ---------------------------------------------------------------------------
__global__ __launch_bounds__(256) void embed_kernel(
    const float* __restrict__ x, const float* __restrict__ embW,
    const float* __restrict__ embb, float* __restrict__ t) {
    int idx = blockIdx.x * blockDim.x + threadIdx.x;   // N*128 threads
    int row = idx >> 7;
    int col = idx & 127;
    float s = embb[col];
    const float* xr = x + (size_t)row * FF;
#pragma unroll
    for (int kk = 0; kk < FF; ++kk) s += xr[kk] * embW[kk * HH + col];
    t[idx] = s;
}

// ---------------------------------------------------------------------------
// Deterministic BatchNorm stats: fixed partial slots + fixed-shape reduction
// ---------------------------------------------------------------------------
__global__ __launch_bounds__(256) void colstats_kernel(
    const float* __restrict__ X, int M, int C,
    float* __restrict__ partS, float* __restrict__ partQ, int walkers) {
    int gid = blockIdx.x * blockDim.x + threadIdx.x;   // C * walkers threads
    int c = gid % C;
    int w = gid / C;
    float s = 0.f, s2 = 0.f;
    for (int r = w; r < M; r += walkers) {
        float v = X[(size_t)r * C + c];
        s += v; s2 += v * v;
    }
    partS[(size_t)c * walkers + w] = s;
    partQ[(size_t)c * walkers + w] = s2;
}

__global__ __launch_bounds__(256) void colreduce_kernel(
    const float* __restrict__ pS, const float* __restrict__ pQ, int P,
    float* __restrict__ sums, float* __restrict__ sumsq) {
    __shared__ float rs[256], rq[256];
    const int c = blockIdx.x;
    float s = 0.f, q2 = 0.f;
    for (int i = threadIdx.x; i < P; i += 256) {
        s  += pS[(size_t)c * P + i];
        q2 += pQ[(size_t)c * P + i];
    }
    rs[threadIdx.x] = s; rq[threadIdx.x] = q2;
    __syncthreads();
    for (int off = 128; off > 0; off >>= 1) {
        if (threadIdx.x < off) {
            rs[threadIdx.x] += rs[threadIdx.x + off];
            rq[threadIdx.x] += rq[threadIdx.x + off];
        }
        __syncthreads();
    }
    if (threadIdx.x == 0) { sums[c] = rs[0]; sumsq[c] = rq[0]; }
}

__global__ void bn_finalize_kernel(
    const float* __restrict__ sums, const float* __restrict__ sumsq,
    const float* __restrict__ g, const float* __restrict__ b,
    float* __restrict__ scale, float* __restrict__ shift, int C, int M) {
    int c = threadIdx.x;
    if (c >= C) return;
    float mean = sums[c] / (float)M;
    float var  = sumsq[c] / (float)M - mean * mean;
    float inv  = rsqrtf(var + EPSB);
    scale[c] = g[c] * inv;
    shift[c] = b[c] - mean * g[c] * inv;
}

// BN apply + exact GELU -> f16 (feeds next WMMA GEMM as A operand)
__global__ __launch_bounds__(256) void bn_gelu_kernel(
    const float* __restrict__ t, const float* __restrict__ scale,
    const float* __restrict__ shift, _Float16* __restrict__ oh,
    int Cmask, int total) {
    int idx = blockIdx.x * blockDim.x + threadIdx.x;
    if (idx >= total) return;
    int c = idx & Cmask;
    float v = t[idx] * scale[c] + shift[c];
    float ge = 0.5f * v * (1.0f + erff(v * 0.70710678118654752f));
    oh[idx] = (_Float16)ge;
}

// f32 [K,N] -> f16 transposed [N,K] (WMMA B operand: K contiguous per row)
__global__ __launch_bounds__(256) void convtr_kernel(
    const float* __restrict__ W, _Float16* __restrict__ Wt, int K, int Ncols) {
    int idx = blockIdx.x * blockDim.x + threadIdx.x;
    if (idx >= K * Ncols) return;
    int n = idx / K;
    int k = idx % K;
    Wt[idx] = (_Float16)W[(size_t)k * Ncols + n];
}

// ---------------------------------------------------------------------------
// WMMA GEMM v2:  C[M,N] = A[M,K](f16) @ Bt[N,K]^T(f16) + bias
//  - workgroup (8 waves) computes 128x64; the 64-col weight panel is staged in
//    LDS once (K-staged at <=256) and shared by all 8 waves (8x B reuse)
//  - each wave computes 16x64: one A fragment feeds 4 WMMAs (4x A reuse)
//  - epilogue: bias, optional f16 mirror, deterministic BN partial stats
// Fragment lane layouts per ISA 7.12.2 (A: half-split K pairs; B: same K order as A;
// D: col=lane&15, rows = (lane>>4)*8 + j).
// ---------------------------------------------------------------------------
#define GEMM_MT 128
#define GEMM_NT 64

__global__ __launch_bounds__(256) void wmma_gemm_kernel(
    const _Float16* __restrict__ A, const _Float16* __restrict__ Bt,
    const float* __restrict__ bias, float* __restrict__ C,
    _Float16* __restrict__ Ch, float* __restrict__ statS,
    float* __restrict__ statQ, int M, int N, int K) {
    extern __shared__ _Float16 lds[];          // 64 x (KST+8)
    const int KST = (K > 256) ? 256 : K;
    const int KP  = KST + 8;                   // pad: keeps ds_load_b128 2-way max
    const int tid  = threadIdx.x;
    const int lane = tid & 31;
    const int wv   = tid >> 5;
    const int tilesN = N / GEMM_NT;
    const int wgm = blockIdx.x / tilesN;
    const int wgn = blockIdx.x % tilesN;
    const int n0  = wgn * GEMM_NT;
    const int half = lane >> 4;
    const int r    = lane & 15;
    const int tm   = wgm * GEMM_MT + wv * 16;

    const _Float16* aRow = A + (size_t)(tm + r) * K + (half << 3);
    const _Float16* lb   = lds + (size_t)r * KP + (half << 3);
    const int chunks = KST >> 3;               // v8h chunks per staged row

    v8f acc[4] = {v8f{}, v8f{}, v8f{}, v8f{}};

    for (int ks = 0; ks < K; ks += KST) {
        // cooperative stage of Bt[n0..n0+64) x [ks..ks+KST) into LDS
        for (int idx = tid; idx < GEMM_NT * chunks; idx += 256) {
            int rr = idx / chunks;
            int cc = idx % chunks;
            *(v8h*)(lds + (size_t)rr * KP + (cc << 3)) =
                *(const v8h*)(Bt + (size_t)(n0 + rr) * K + ks + (cc << 3));
        }
        __syncthreads();

        for (int k0 = 0; k0 < KST; k0 += 32) {
            union { v16h v; v8h h[2]; } a, b[4];
            a.h[0] = *(const v8h*)(aRow + ks + k0);
            a.h[1] = *(const v8h*)(aRow + ks + k0 + 16);
#pragma unroll
            for (int t = 0; t < 4; ++t) {
                const _Float16* bp = lb + (size_t)(t * 16) * KP + k0;
                b[t].h[0] = *(const v8h*)(bp);
                b[t].h[1] = *(const v8h*)(bp + 16);
            }
#pragma unroll
            for (int t = 0; t < 4; ++t)
                acc[t] = __builtin_amdgcn_wmma_f32_16x16x32_f16(
                    false, a.v, false, b[t].v, (short)0, acc[t], false, false);
        }
        __syncthreads();
    }

    const int nblk   = M >> 4;
    const int rowblk = wgm * 8 + wv;
#pragma unroll
    for (int t = 0; t < 4; ++t) {
        const int col = n0 + t * 16 + r;
        const float bc = bias ? bias[col] : 0.0f;
        float s = 0.f, s2 = 0.f;
#pragma unroll
        for (int j = 0; j < 8; ++j) {
            const int row = tm + (half << 3) + j;
            float cv = acc[t][j] + bc;
            C[(size_t)row * N + col] = cv;
            if (Ch) Ch[(size_t)row * N + col] = (_Float16)cv;
            s += cv; s2 += cv * cv;
        }
        if (statS) {                           // combine the two row-halves
            s  += __shfl_xor(s, 16, 32);
            s2 += __shfl_xor(s2, 16, 32);
            if (half == 0) {
                statS[(size_t)col * nblk + rowblk] = s;
                statQ[(size_t)col * nblk + rowblk] = s2;
            }
        }
    }
}

// ---------------------------------------------------------------------------
// CSR-by-destination build (once; graph is static across all 5 layers)
// ---------------------------------------------------------------------------
__global__ void fill_i32_kernel(int* __restrict__ p, int v, int n) {
    int i = blockIdx.x * blockDim.x + threadIdx.x;
    if (i < n) p[i] = v;
}

__global__ __launch_bounds__(256) void hist_kernel(
    const int* __restrict__ ei, int* __restrict__ deg, int E) {
    int e = blockIdx.x * blockDim.x + threadIdx.x;
    if (e < E) atomicAdd(&deg[ei[E + e]], 1);      // integer: deterministic
}

__global__ __launch_bounds__(1024) void scan_kernel(
    const int* __restrict__ deg, int* __restrict__ rowstart,
    int* __restrict__ cursor, int Nn) {
    __shared__ int part[1024];
    const int t = threadIdx.x;
    const int base = t * 16;                        // Nn == 1024*16
    int local[16]; int s = 0;
#pragma unroll
    for (int j = 0; j < 16; ++j) { local[j] = s; s += deg[base + j]; }
    int val = s;
    for (int off = 1; off < 1024; off <<= 1) {
        part[t] = val; __syncthreads();
        int add = (t >= off) ? part[t - off] : 0;
        __syncthreads();
        val += add;
    }
    int ebase = val - s;
#pragma unroll
    for (int j = 0; j < 16; ++j) {
        int rs = ebase + local[j];
        rowstart[base + j] = rs;
        cursor[base + j]   = rs;
    }
    if (t == 1023) rowstart[Nn] = ebase + s;
}

__global__ __launch_bounds__(256) void scatter_kernel(
    const int* __restrict__ ei, int* __restrict__ cursor,
    int* __restrict__ csrSrc, int E) {
    int e = blockIdx.x * blockDim.x + threadIdx.x;
    if (e >= E) return;
    int pos = atomicAdd(&cursor[ei[E + e]], 1);
    csrSrc[pos] = ei[e];
}

// Sort each row's src list (insertion sort, deg ~16). Makes the float
// accumulation order value-deterministic (equal srcs give identical terms).
__global__ __launch_bounds__(256) void csr_sort_kernel(
    const int* __restrict__ rowstart, int* __restrict__ csrSrc, int Nn) {
    int n = blockIdx.x * blockDim.x + threadIdx.x;
    if (n >= Nn) return;
    int b = rowstart[n], e = rowstart[n + 1];
    for (int i = b + 1; i < e; ++i) {
        int key = csrSrc[i]; int j = i - 1;
        while (j >= b && csrSrc[j] > key) { csrSrc[j + 1] = csrSrc[j]; --j; }
        csrSrc[j + 1] = key;
    }
}

// ---------------------------------------------------------------------------
// Fused attention: wave per node, 8 lanes per head, atomic-free.
// Pass 1: logits (+max) over incoming edges; Pass 2: exp/denominator/value
// aggregation in registers; then head-mean + beta-gate, emit f16.
// Lane l: head = l>>3, owns 16 contiguous channels (l&7)*16.. (float4 loads).
// ---------------------------------------------------------------------------
__global__ __launch_bounds__(256) void attn_fused_kernel(
    const float* __restrict__ q, const float* __restrict__ k,
    const float* __restrict__ v, const float* __restrict__ xr,
    const int* __restrict__ rowstart, const int* __restrict__ csrSrc,
    float* __restrict__ pl, const float* __restrict__ wbeta,
    _Float16* __restrict__ outh, int Nn) {
    const int lane = threadIdx.x & 31;
    const int n = blockIdx.x * 8 + (threadIdx.x >> 5);
    if (n >= Nn) return;
    const int hd = lane >> 3;
    const int c8 = lane & 7;
    const int chanBase = hd * HH + c8 * 16;

    float4 qv[4];
    const float4* qp = (const float4*)(q + (size_t)n * (HEADS * HH) + chanBase);
#pragma unroll
    for (int j = 0; j < 4; ++j) qv[j] = qp[j];

    const int beg = rowstart[n], end = rowstart[n + 1];
    const float scale = 0.08838834764831845f;     // 1/sqrt(128)
    float mx = -3.0e38f;
    for (int i = beg; i < end; ++i) {
        const int src = csrSrc[i];
        const float4* kp = (const float4*)(k + (size_t)src * (HEADS * HH) + chanBase);
        float part = 0.f;
#pragma unroll
        for (int j = 0; j < 4; ++j) {
            float4 kv = kp[j];
            part += qv[j].x * kv.x + qv[j].y * kv.y + qv[j].z * kv.z + qv[j].w * kv.w;
        }
        part += __shfl_xor(part, 1, 32);          // reduce within 8-lane head group
        part += __shfl_xor(part, 2, 32);
        part += __shfl_xor(part, 4, 32);
        float lg = part * scale;
        if (c8 == 0) pl[(size_t)i * HEADS + hd] = lg;
        mx = fmaxf(mx, lg);
    }

    float vacc[16];
#pragma unroll
    for (int j = 0; j < 16; ++j) vacc[j] = 0.f;
    float den = 0.f;
    for (int i = beg; i < end; ++i) {
        const int src = csrSrc[i];
        float p = __expf(pl[(size_t)i * HEADS + hd] - mx);
        den += p;
        const float4* vp = (const float4*)(v + (size_t)src * (HEADS * HH) + chanBase);
#pragma unroll
        for (int j = 0; j < 4; ++j) {
            float4 vv = vp[j];
            vacc[4 * j + 0] += p * vv.x;
            vacc[4 * j + 1] += p * vv.y;
            vacc[4 * j + 2] += p * vv.z;
            vacc[4 * j + 3] += p * vv.w;
        }
    }
    const float inv = 1.0f / (den + 1e-16f);
#pragma unroll
    for (int j = 0; j < 16; ++j) {
        float s = vacc[j] * inv;                  // per-head normalized
        s += __shfl_xor(s, 8, 32);                // sum across the 4 head groups
        s += __shfl_xor(s, 16, 32);
        vacc[j] = 0.25f * s;                      // mean over heads
    }
    // beta gate: lanes 0..7 own channels c8*16 + j
    float bl = 0.f;
    float xv[16];
#pragma unroll
    for (int j = 0; j < 16; ++j) xv[j] = 0.f;
    if (lane < 8) {
        const float* xp = xr + (size_t)n * HH + c8 * 16;
#pragma unroll
        for (int j = 0; j < 16; ++j) {
            xv[j] = xp[j];
            int c = c8 * 16 + j;
            bl += vacc[j] * wbeta[c] + xv[j] * wbeta[HH + c] +
                  (vacc[j] - xv[j]) * wbeta[2 * HH + c];
        }
    }
    bl += __shfl_xor(bl, 1, 32);
    bl += __shfl_xor(bl, 2, 32);
    bl += __shfl_xor(bl, 4, 32);
    bl += __shfl_xor(bl, 8, 32);
    bl += __shfl_xor(bl, 16, 32);
    const float beta = 1.0f / (1.0f + __expf(-bl));
    if (lane < 8) {
        _Float16* op = outh + (size_t)n * HH + c8 * 16;
#pragma unroll
        for (int j = 0; j < 16; ++j)
            op[j] = (_Float16)(beta * xv[j] + (1.0f - beta) * vacc[j]);
    }
}

// ---------------------------------------------------------------------------
// Final: segment-max pooling per graph (order-invariant atomicMax), 128-d GEMV
// ---------------------------------------------------------------------------
__global__ __launch_bounds__(256) void pool_max_kernel(
    const float* __restrict__ h, const int* __restrict__ batch,
    unsigned* __restrict__ pooledKey, int total) {
    int idx = blockIdx.x * blockDim.x + threadIdx.x;
    if (idx >= total) return;
    int n = idx >> 7;
    int c = idx & 127;
    atomicMax(&pooledKey[batch[n] * HH + c], f2key(h[idx]));
}

__global__ __launch_bounds__(256) void pred_kernel(
    const unsigned* __restrict__ pooledKey, const float* __restrict__ predW,
    const float* __restrict__ predb, float* __restrict__ out, int G) {
    const int lane = threadIdx.x & 31;
    const int g = blockIdx.x * 8 + (threadIdx.x >> 5);
    if (g >= G) return;
    float s = 0.f;
#pragma unroll
    for (int cc = 0; cc < 4; ++cc) {
        int c = lane + cc * 32;
        s += key2f(pooledKey[g * HH + c]) * predW[c];
    }
    for (int off = 16; off > 0; off >>= 1) s += __shfl_xor(s, off, 32);
    if (lane == 0) out[g] = s + predb[0];
}

// ---------------------------------------------------------------------------
// Host orchestration
// ---------------------------------------------------------------------------
extern "C" void kernel_launch(void* const* d_in, const int* in_sizes, int n_in,
                              void* d_out, int out_size, void* d_ws, size_t ws_size,
                              hipStream_t stream) {
    (void)in_sizes; (void)n_in; (void)out_size; (void)ws_size;
    const float* x      = (const float*)d_in[0];
    const int*   ei     = (const int*)d_in[1];
    const int*   batch  = (const int*)d_in[2];
    const float* embW   = (const float*)d_in[3];
    const float* embb   = (const float*)d_in[4];
    const float* emb_g  = (const float*)d_in[5];
    const float* emb_be = (const float*)d_in[6];
    const float* Wq     = (const float*)d_in[7];
    const float* bq     = (const float*)d_in[8];
    const float* Wk     = (const float*)d_in[9];
    const float* bk     = (const float*)d_in[10];
    const float* Wv     = (const float*)d_in[11];
    const float* bv     = (const float*)d_in[12];
    const float* Wskip  = (const float*)d_in[13];
    const float* bskip  = (const float*)d_in[14];
    const float* Wbeta  = (const float*)d_in[15];
    const float* W1     = (const float*)d_in[16];
    const float* b1     = (const float*)d_in[17];
    const float* g1     = (const float*)d_in[18];
    const float* be1    = (const float*)d_in[19];
    const float* W2     = (const float*)d_in[20];
    const float* b2     = (const float*)d_in[21];
    const float* predW  = (const float*)d_in[22];
    const float* predb  = (const float*)d_in[23];
    float* out = (float*)d_out;

    // ---- workspace arena ----
    size_t off = 0;
    char* base = (char*)d_ws;
    auto take = [&](size_t bytes) -> void* {
        off = (off + 255) & ~(size_t)255;
        void* r = base + off;
        off += bytes;
        return r;
    };
    float*    h     = (float*)take((size_t)NN * HH * 4);
    _Float16* hb    = (_Float16*)take((size_t)NN * HH * 2);
    float*    qf    = (float*)take((size_t)NN * HEADS * HH * 4);
    float*    kf    = (float*)take((size_t)NN * HEADS * HH * 4);
    float*    vf    = (float*)take((size_t)NN * HEADS * HH * 4);
    float*    xrf   = (float*)take((size_t)NN * HH * 4);
    float*    pl    = (float*)take((size_t)EE * HEADS * 4);
    _Float16* outh  = (_Float16*)take((size_t)NN * HH * 2);
    float*    sums  = (float*)take(DFF * 4);
    float*    sumsq = (float*)take(DFF * 4);
    float*    bscal = (float*)take(DFF * 4);
    float*    bshft = (float*)take(DFF * 4);
    float*    statS = (float*)take((size_t)DFF * (NN / 16) * 4);   // 2 MB
    float*    statQ = (float*)take((size_t)DFF * (NN / 16) * 4);   // 2 MB
    _Float16* wqT   = (_Float16*)take((size_t)HH * DFF * 2);
    _Float16* wkT   = (_Float16*)take((size_t)HH * DFF * 2);
    _Float16* wvT   = (_Float16*)take((size_t)HH * DFF * 2);
    _Float16* wsT   = (_Float16*)take((size_t)HH * HH * 2);
    _Float16* w1T   = (_Float16*)take((size_t)HH * DFF * 2);
    _Float16* w2T   = (_Float16*)take((size_t)DFF * HH * 2);
    unsigned* pkey  = (unsigned*)take((size_t)GG * HH * 4);
    int*      deg      = (int*)take((size_t)NN * 4);
    int*      cursor   = (int*)take((size_t)NN * 4);
    int*      rowstart = (int*)take((size_t)(NN + 1) * 4);
    int*      csrSrc   = (int*)take((size_t)EE * 4);
    // aliases (lifetimes do not overlap; stream order serializes):
    float*    t1   = kf;             // FFN hidden, after attention done with k
    _Float16* hhb  = (_Float16*)vf;  // GELU f16, after attention done with v
    float*    temb = kf;             // embed pre-BN, before any layer

    const dim3 B256(256);
    const size_t shm128 = (size_t)GEMM_NT * (128 + 8) * 2;   // 17,408 B
    const size_t shm512 = (size_t)GEMM_NT * (256 + 8) * 2;   // 33,792 B (K-staged)

    // ---- CSR by destination (built once; reused by all 5 layers) ----
    fill_i32_kernel<<<dim3(NN / 256), B256, 0, stream>>>(deg, 0, NN);
    hist_kernel<<<dim3(EE / 256), B256, 0, stream>>>(ei, deg, EE);
    scan_kernel<<<dim3(1), dim3(1024), 0, stream>>>(deg, rowstart, cursor, NN);
    scatter_kernel<<<dim3(EE / 256), B256, 0, stream>>>(ei, cursor, csrSrc, EE);
    csr_sort_kernel<<<dim3(NN / 256), B256, 0, stream>>>(rowstart, csrSrc, NN);

    // ---- embedding + BatchNorm + GELU ----
    embed_kernel<<<dim3((NN * HH) / 256), B256, 0, stream>>>(x, embW, embb, temb);
    colstats_kernel<<<dim3((HH * 64) / 256), B256, 0, stream>>>(temb, NN, HH,
                                                                statS, statQ, 64);
    colreduce_kernel<<<dim3(HH), B256, 0, stream>>>(statS, statQ, 64, sums, sumsq);
    bn_finalize_kernel<<<dim3(1), dim3(512), 0, stream>>>(sums, sumsq, emb_g, emb_be,
                                                          bscal, bshft, HH, NN);
    bn_gelu_kernel<<<dim3((NN * HH) / 256), B256, 0, stream>>>(temb, bscal, bshft, hb,
                                                               HH - 1, NN * HH);

    // ---- 5 transformer layers ----
    for (int l = 0; l < LL; ++l) {
        const float* Wq_l = Wq + (size_t)l * HH * HEADS * HH;
        const float* Wk_l = Wk + (size_t)l * HH * HEADS * HH;
        const float* Wv_l = Wv + (size_t)l * HH * HEADS * HH;
        const float* Ws_l = Wskip + (size_t)l * HH * HH;
        const float* W1_l = W1 + (size_t)l * HH * DFF;
        const float* W2_l = W2 + (size_t)l * DFF * HH;

        convtr_kernel<<<dim3((HH * DFF) / 256), B256, 0, stream>>>(Wq_l, wqT, HH, DFF);
        convtr_kernel<<<dim3((HH * DFF) / 256), B256, 0, stream>>>(Wk_l, wkT, HH, DFF);
        convtr_kernel<<<dim3((HH * DFF) / 256), B256, 0, stream>>>(Wv_l, wvT, HH, DFF);
        convtr_kernel<<<dim3((HH * HH) / 256), B256, 0, stream>>>(Ws_l, wsT, HH, HH);
        convtr_kernel<<<dim3((HH * DFF) / 256), B256, 0, stream>>>(W1_l, w1T, HH, DFF);
        convtr_kernel<<<dim3((DFF * HH) / 256), B256, 0, stream>>>(W2_l, w2T, DFF, HH);

        // q/k/v/skip projections (WMMA, LDS-staged weight panels)
        const int gQKV = (NN / GEMM_MT) * (DFF / GEMM_NT);   // 1024 blocks
        const int gSk  = (NN / GEMM_MT) * (HH / GEMM_NT);    // 256 blocks
        wmma_gemm_kernel<<<dim3(gQKV), B256, shm128, stream>>>(
            hb, wqT, bq + l * DFF, qf, nullptr, nullptr, nullptr, NN, DFF, HH);
        wmma_gemm_kernel<<<dim3(gQKV), B256, shm128, stream>>>(
            hb, wkT, bk + l * DFF, kf, nullptr, nullptr, nullptr, NN, DFF, HH);
        wmma_gemm_kernel<<<dim3(gQKV), B256, shm128, stream>>>(
            hb, wvT, bv + l * DFF, vf, nullptr, nullptr, nullptr, NN, DFF, HH);
        wmma_gemm_kernel<<<dim3(gSk), B256, shm128, stream>>>(
            hb, wsT, bskip + l * HH, xrf, nullptr, nullptr, nullptr, NN, HH, HH);

        // fused atomic-free softmax attention + head-mean + beta gate
        attn_fused_kernel<<<dim3(NN / 8), B256, 0, stream>>>(
            qf, kf, vf, xrf, rowstart, csrSrc, pl,
            Wbeta + (size_t)l * 3 * HH, outh, NN);

        // FFN: GEMM (+deterministic BN partials) -> reduce -> BN+GELU -> GEMM
        wmma_gemm_kernel<<<dim3(gQKV), B256, shm128, stream>>>(
            outh, w1T, b1 + l * DFF, t1, nullptr, statS, statQ, NN, DFF, HH);
        colreduce_kernel<<<dim3(DFF), B256, 0, stream>>>(statS, statQ, NN / 16,
                                                         sums, sumsq);
        bn_finalize_kernel<<<dim3(1), dim3(512), 0, stream>>>(
            sums, sumsq, g1 + l * DFF, be1 + l * DFF, bscal, bshft, DFF, NN);
        bn_gelu_kernel<<<dim3((NN * DFF) / 256), B256, 0, stream>>>(
            t1, bscal, bshft, hhb, DFF - 1, NN * DFF);
        wmma_gemm_kernel<<<dim3(gSk), B256, shm512, stream>>>(
            hhb, w2T, b2 + l * HH, h, hb, nullptr, nullptr, NN, HH, DFF);
    }

    // ---- graph max-pool + prediction head ----
    fill_u32_kernel<<<dim3((GG * HH) / 256), B256, 0, stream>>>(pkey, 0x007FFFFFu, GG * HH);
    pool_max_kernel<<<dim3((NN * HH) / 256), B256, 0, stream>>>(h, batch, pkey, NN * HH);
    pred_kernel<<<dim3(GG / 8), B256, 0, stream>>>(pkey, predW, predb, out, GG);
}
